// DockerBlock_39659728011893
// MI455X (gfx1250) — matrix operand order repaired and checked by reading, not compiled, not run on hardware
//
#include <hip/hip_runtime.h>
#include <hip/hip_bf16.h>
#include <math.h>

typedef __bf16 bf16_t;
typedef __attribute__((ext_vector_type(16))) __bf16 v16bf;
typedef __attribute__((ext_vector_type(8)))  float        v8f;
typedef __attribute__((ext_vector_type(8)))  unsigned int v8u;

#define NRES 768
#define DIMC 384
#define EDIMC 128
#define HNUM 8

static constexpr float SCALAR_SCALE_C = 0.14433756729740643f; // (3*16)^-0.5
static constexpr float POINT_SCALE_C  = 0.13608276348795434f; // (3*4*4.5)^-0.5
static constexpr float PAIR_SCALE_C   = 0.57735026918962576f; // 3^-0.5

__device__ __forceinline__ bf16_t f2bf(float f) {
    unsigned u = __float_as_uint(f);
    u += ((u >> 16) & 1u) + 0x7fffu;            // round-to-nearest-even
    unsigned short s = (unsigned short)(u >> 16);
    return __builtin_bit_cast(bf16_t, s);
}
__device__ __forceinline__ bf16_t zbf() {
    unsigned short s = 0;
    return __builtin_bit_cast(bf16_t, s);
}

// ---------------------------------------------------------------------------
// Generic batched bf16 WMMA GEMM:  C = alpha*(A @ Bt^T) [+ C] [+ bias] [relu]
// A : [M,K] bf16 row-major (row stride a_rs, batch stride a_bs)
// Bt: [N,K] bf16 row-major (B transposed; row stride b_rs, batch stride b_bs)
// C : [M,N] f32 (row stride c_rs, batch stride c_bs); optional bf16 copy Cbf.
// M,N multiples of 16; K multiple of 32. One wave per 16x16 C tile.
// ---------------------------------------------------------------------------
__global__ __launch_bounds__(128) void k_gemm_bf16(
    const bf16_t* __restrict__ A, long long a_rs, long long a_bs,
    const bf16_t* __restrict__ Bt, long long b_rs, long long b_bs,
    float* __restrict__ C, long long c_rs, long long c_bs,
    bf16_t* __restrict__ Cbf,
    const float* __restrict__ bias,
    int M, int N, int K, float alpha, int acc, int relu)
{
    int wave = threadIdx.x >> 5;
    int lane = threadIdx.x & 31;
    int tiles_n = N >> 4;
    int tiles_m = M >> 4;
    int tile = blockIdx.x * 4 + wave;
    if (tile >= tiles_m * tiles_n) return;         // wave-uniform
    int tm = tile / tiles_n, tn = tile % tiles_n;

    long long batch = blockIdx.y;
    const bf16_t* Ab = A + batch * a_bs;
    const bf16_t* Bb = Bt + batch * b_bs;
    float*        Cb = C + batch * c_bs;

    int mlane = lane & 15;
    int hi    = lane >> 4;
    int akb   = hi ? 16 : 0;   // A lane K-base (contiguous K, matches B)
    int bkb   = hi ? 16 : 0;   // B lane K-base (contiguous K, matches A)
    const bf16_t* arow = Ab + (long long)(tm * 16 + mlane) * a_rs;
    const bf16_t* brow = Bb + (long long)(tn * 16 + mlane) * b_rs;

    v8f c = {0.f,0.f,0.f,0.f,0.f,0.f,0.f,0.f};
    for (int k0 = 0; k0 < K; k0 += 32) {
        v8u au, bu;
        #pragma unroll
        for (int v = 0; v < 8; ++v) {
            int ka = akb + 2 * v;
            au[v] = *(const unsigned int*)(arow + k0 + ka);
            bu[v] = *(const unsigned int*)(brow + k0 + bkb + 2 * v);
        }
        v16bf av = __builtin_bit_cast(v16bf, au);
        v16bf bv = __builtin_bit_cast(v16bf, bu);
        c = __builtin_amdgcn_wmma_f32_16x16x32_bf16(false, av, false, bv,
                                                    (short)0, c, false, false);
    }
    int n = tn * 16 + mlane;
    float bv_ = bias ? bias[n] : 0.0f;
    #pragma unroll
    for (int v = 0; v < 8; ++v) {
        int m = tm * 16 + v + (hi ? 8 : 0);        // C layout: V_v = rows v / v+8
        long long off = (long long)m * c_rs + n;
        float x = c[v] * alpha;
        if (acc) x += Cb[off];
        x += bv_;
        if (relu) x = fmaxf(x, 0.0f);
        Cb[off] = x;
        if (Cbf) Cbf[batch * c_bs + off] = f2bf(x);
    }
}

// ---------------------------------------------------------------------------
// Pass 1 over edges: pair-bias logits init.
// logits[h][i][j] = PAIR_SCALE * (edges[i][j][:] . Wpair[:,h] + bpair[h])
// ---------------------------------------------------------------------------
__global__ __launch_bounds__(256) void k_pair_logits(
    const float* __restrict__ edges, const float* __restrict__ Wpair,
    const float* __restrict__ bpair, float* __restrict__ logits)
{
    __shared__ float e[64][129];
    __shared__ float w[128][8];
    int i = blockIdx.y;
    int j0 = blockIdx.x * 64;
    int t = threadIdx.x;
    for (int idx = t; idx < 128 * 8; idx += 256) w[idx >> 3][idx & 7] = Wpair[idx];
    #pragma unroll
    for (int it = 0; it < 8; ++it) {
        int idx = t + it * 256;                 // 0..2047 float4
        int jj = idx >> 5;
        int d4 = (idx & 31) * 4;
        float4 v = *(const float4*)(edges + ((size_t)i * NRES + j0 + jj) * EDIMC + d4);
        e[jj][d4 + 0] = v.x; e[jj][d4 + 1] = v.y;
        e[jj][d4 + 2] = v.z; e[jj][d4 + 3] = v.w;
    }
    __syncthreads();
    #pragma unroll
    for (int rep = 0; rep < 2; ++rep) {
        int idx = t + rep * 256;
        int jj = idx & 63;
        int h  = idx >> 6;
        float s = 0.f;
        for (int d = 0; d < 128; ++d) s += e[jj][d] * w[d][h];
        logits[((size_t)h * NRES + i) * NRES + j0 + jj] = PAIR_SCALE_C * (s + bpair[h]);
    }
}

// ---------------------------------------------------------------------------
// Point-distance logit term: logits -= 0.5*POINT_SCALE*softplus(pw[h]) * dist
// ---------------------------------------------------------------------------
__global__ void k_point_logits(const float* __restrict__ qp, const float* __restrict__ kp,
                               const float* __restrict__ pw, float* __restrict__ logits)
{
    int j = blockIdx.x * 256 + threadIdx.x;
    int i = blockIdx.y, h = blockIdx.z;
    if (j >= NRES) return;
    float coef = 0.5f * POINT_SCALE_C * log1pf(expf(pw[h]));
    const float* q = qp + ((size_t)h * NRES + i) * 12;
    const float* k = kp + ((size_t)h * NRES + j) * 12;
    float dist = 0.f;
    #pragma unroll
    for (int d = 0; d < 4; ++d) {
        float dx = q[d*3+0]-k[d*3+0], dy = q[d*3+1]-k[d*3+1], dz = q[d*3+2]-k[d*3+2];
        dist += dx*dx + dy*dy + dz*dz;
    }
    logits[((size_t)h * NRES + i) * NRES + j] -= coef * dist;
}

// ---------------------------------------------------------------------------
// Masked softmax over j; writes attn in bf16 (all consumers are WMMA).
// ---------------------------------------------------------------------------
__global__ __launch_bounds__(256) void k_softmax(const float* __restrict__ logits,
                                                 const unsigned char* __restrict__ mask,
                                                 bf16_t* __restrict__ attn)
{
    __shared__ float sm[256];
    int i = blockIdx.x, h = blockIdx.y, t = threadIdx.x;
    const float* row = logits + ((size_t)h * NRES + i) * NRES;
    bool mi = mask[i] != 0;
    float v0[3], mx = -3.0e38f;
    #pragma unroll
    for (int r = 0; r < 3; ++r) {
        int j = t + r * 256;
        float v = (mi && (mask[j] != 0)) ? row[j] : -3.0e38f;
        v0[r] = v; mx = fmaxf(mx, v);
    }
    sm[t] = mx; __syncthreads();
    for (int o = 128; o; o >>= 1) { if (t < o) sm[t] = fmaxf(sm[t], sm[t + o]); __syncthreads(); }
    mx = sm[0]; __syncthreads();
    float s = 0.f, e0[3];
    #pragma unroll
    for (int r = 0; r < 3; ++r) { float e = expf(v0[r] - mx); e0[r] = e; s += e; }
    sm[t] = s; __syncthreads();
    for (int o = 128; o; o >>= 1) { if (t < o) sm[t] += sm[t + o]; __syncthreads(); }
    float inv = 1.0f / sm[0];
    bf16_t* arow = attn + ((size_t)h * NRES + i) * NRES;
    #pragma unroll
    for (int r = 0; r < 3; ++r) arow[t + r * 256] = f2bf(e0[r] * inv);
}

// ---------------------------------------------------------------------------
// Pass 2 over edges (specialized WMMA): res_pairT[i][d][h16] = attn[h,i,:] . edges[i,:,d]
// One workgroup (8 waves) per i; M=d (wave*16), N=h (16, h>=8 discarded), K=j.
// edges streamed f32 -> bf16 through LDS; A operand built from LDS, contiguous K (matches B).
// ---------------------------------------------------------------------------
__global__ __launch_bounds__(256) void k_res_pair(const float* __restrict__ edges,
                                                  const bf16_t* __restrict__ attn,
                                                  float* __restrict__ rpt)
{
    __shared__ bf16_t eb[32][132];
    int i = blockIdx.x;
    int t = threadIdx.x;
    int wave = t >> 5, lane = t & 31;
    int mlane = lane & 15, hi = lane >> 4;
    int akb = hi ? 16 : 0, bkb = hi ? 16 : 0;
    const bf16_t* brow = attn + (size_t)mlane * (NRES * (size_t)NRES) + (size_t)i * NRES;
    int d = wave * 16 + mlane;
    v8f c = {0.f,0.f,0.f,0.f,0.f,0.f,0.f,0.f};
    for (int j0 = 0; j0 < NRES; j0 += 32) {
        __syncthreads();
        #pragma unroll
        for (int it = 0; it < 4; ++it) {
            int jj = (t >> 5) + it * 8;          // 0..31
            int d4 = (t & 31) * 4;
            float4 v = *(const float4*)(edges + ((size_t)i * NRES + j0 + jj) * EDIMC + d4);
            eb[jj][d4 + 0] = f2bf(v.x); eb[jj][d4 + 1] = f2bf(v.y);
            eb[jj][d4 + 2] = f2bf(v.z); eb[jj][d4 + 3] = f2bf(v.w);
        }
        __syncthreads();
        v8u au, bu;
        #pragma unroll
        for (int v = 0; v < 8; ++v) {
            int ka = akb + 2 * v;
            unsigned lo  = __builtin_bit_cast(unsigned short, eb[ka][d]);
            unsigned hi2 = __builtin_bit_cast(unsigned short, eb[ka + 1][d]);
            au[v] = lo | (hi2 << 16);
            bu[v] = *(const unsigned int*)(brow + j0 + bkb + 2 * v);
        }
        v16bf av = __builtin_bit_cast(v16bf, au);
        v16bf bv = __builtin_bit_cast(v16bf, bu);
        c = __builtin_amdgcn_wmma_f32_16x16x32_bf16(false, av, false, bv,
                                                    (short)0, c, false, false);
    }
    #pragma unroll
    for (int v = 0; v < 8; ++v) {
        int m = wave * 16 + v + (hi ? 8 : 0);     // d index
        rpt[((size_t)i * 128 + m) * 16 + mlane] = c[v];
    }
}

// --------------------------- small helper kernels ---------------------------
__global__ void k_f32_to_bf16(const float* __restrict__ s, bf16_t* __restrict__ d, int n) {
    int i = blockIdx.x * 256 + threadIdx.x;
    if (i < n) d[i] = f2bf(s[i]);
}

// dst[n][k] (bf16, Npad rows) = src[k][n] (f32 [K][Nv]); zero-pad rows >= Nv
__global__ void k_wtrans(const float* __restrict__ src, bf16_t* __restrict__ dst,
                         int K, int Nv, int Npad) {
    int tid = blockIdx.x * 256 + threadIdx.x;
    if (tid >= Npad * K) return;
    int n = tid / K, k = tid - n * K;
    dst[tid] = (n < Nv) ? f2bf(src[(size_t)k * Nv + n]) : zbf();
}

__global__ void k_padbias(const float* __restrict__ s, float* __restrict__ d, int nv, int np) {
    int t = threadIdx.x;
    if (t < np) d[t] = (t < nv) ? s[t] : 0.0f;
}

// qs/ks/vs f32 [n][H*16] -> qsb/ksb bf16 [h][n][32] (K-padded), vsT bf16 [h][16][n]
__global__ void k_pack_qkv(const float* __restrict__ qs, const float* __restrict__ ks,
                           const float* __restrict__ vs,
                           bf16_t* __restrict__ qsb, bf16_t* __restrict__ ksb,
                           bf16_t* __restrict__ vsT) {
    int tid = blockIdx.x * 256 + threadIdx.x;
    if (tid >= HNUM * NRES * 32) return;
    int c = tid & 31;
    int n = (tid >> 5) % NRES;
    int h = tid / (NRES * 32);
    if (c < 16) {
        size_t src = (size_t)n * 128 + h * 16 + c;
        qsb[tid] = f2bf(qs[src]);
        ksb[tid] = f2bf(ks[src]);
        vsT[((size_t)h * 16 + c) * NRES + n] = f2bf(vs[src]);
    } else {
        qsb[tid] = zbf();
        ksb[tid] = zbf();
    }
}

// Apply frames: qp/kp[h][n][12] = R(n)^T-form transform + t ; vpt bf16 [h][32][n]
__global__ void k_points(const float* __restrict__ qpr, const float* __restrict__ kpr,
                         const float* __restrict__ vpr, const float* __restrict__ rot,
                         const float* __restrict__ coors,
                         float* __restrict__ qp, float* __restrict__ kp,
                         bf16_t* __restrict__ vpt) {
    int n = blockIdx.x * 256 + threadIdx.x;
    if (n >= NRES) return;
    float R[9];
    #pragma unroll
    for (int a = 0; a < 9; ++a) R[a] = rot[(size_t)n * 9 + a];
    float t0 = coors[n*3], t1 = coors[n*3+1], t2 = coors[n*3+2];
    for (int h = 0; h < HNUM; ++h) {
        #pragma unroll
        for (int d = 0; d < 4; ++d) {
            const float* pq = qpr + (size_t)n * 96 + h * 12 + d * 3;
            const float* pk = kpr + (size_t)n * 96 + h * 12 + d * 3;
            float* oq = qp + ((size_t)h * NRES + n) * 12 + d * 3;
            float* ok = kp + ((size_t)h * NRES + n) * 12 + d * 3;
            // out[r] = sum_c p[c]*R[c][r] + t[r]
            oq[0] = pq[0]*R[0] + pq[1]*R[3] + pq[2]*R[6] + t0;
            oq[1] = pq[0]*R[1] + pq[1]*R[4] + pq[2]*R[7] + t1;
            oq[2] = pq[0]*R[2] + pq[1]*R[5] + pq[2]*R[8] + t2;
            ok[0] = pk[0]*R[0] + pk[1]*R[3] + pk[2]*R[6] + t0;
            ok[1] = pk[0]*R[1] + pk[1]*R[4] + pk[2]*R[7] + t1;
            ok[2] = pk[0]*R[2] + pk[1]*R[5] + pk[2]*R[8] + t2;
        }
        #pragma unroll
        for (int d = 0; d < 8; ++d) {
            const float* pv = vpr + (size_t)n * 192 + h * 24 + d * 3;
            float o0 = pv[0]*R[0] + pv[1]*R[3] + pv[2]*R[6] + t0;
            float o1 = pv[0]*R[1] + pv[1]*R[4] + pv[2]*R[7] + t1;
            float o2 = pv[0]*R[2] + pv[1]*R[5] + pv[2]*R[8] + t2;
            vpt[((size_t)h * 32 + d*3 + 0) * NRES + n] = f2bf(o0);
            vpt[((size_t)h * 32 + d*3 + 1) * NRES + n] = f2bf(o1);
            vpt[((size_t)h * 32 + d*3 + 2) * NRES + n] = f2bf(o2);
        }
        #pragma unroll
        for (int cpad = 24; cpad < 32; ++cpad)
            vpt[((size_t)h * 32 + cpad) * NRES + n] = zbf();
    }
}

// Inverse frame on res_p, plus norms.
__global__ void k_resp_local(const float* __restrict__ rpg, const float* __restrict__ rot,
                             const float* __restrict__ coors,
                             float* __restrict__ rpl, float* __restrict__ rpn) {
    int tid = blockIdx.x * 256 + threadIdx.x;
    if (tid >= HNUM * NRES) return;
    int n = tid % NRES;
    float R[9];
    #pragma unroll
    for (int a = 0; a < 9; ++a) R[a] = rot[(size_t)n * 9 + a];
    float t0 = coors[n*3], t1 = coors[n*3+1], t2 = coors[n*3+2];
    const float* g = rpg + (size_t)tid * 32;
    #pragma unroll
    for (int d = 0; d < 8; ++d) {
        float x = g[d*3+0] - t0, y = g[d*3+1] - t1, z = g[d*3+2] - t2;
        // local[r] = sum_c v[c]*R[r][c]
        float l0 = x*R[0] + y*R[1] + z*R[2];
        float l1 = x*R[3] + y*R[4] + z*R[5];
        float l2 = x*R[6] + y*R[7] + z*R[8];
        rpl[(size_t)tid * 24 + d*3 + 0] = l0;
        rpl[(size_t)tid * 24 + d*3 + 1] = l1;
        rpl[(size_t)tid * 24 + d*3 + 2] = l2;
        rpn[(size_t)tid * 8 + d] = sqrtf(l0*l0 + l1*l1 + l2*l2 + 1e-8f);
    }
}

// Concat [res_s | res_p | res_pn | res_pair] -> bf16 [768][1408]
__global__ void k_concat(const float* __restrict__ res_s, const float* __restrict__ rpl,
                         const float* __restrict__ rpn, const float* __restrict__ rpt,
                         bf16_t* __restrict__ of) {
    int tid = blockIdx.x * 256 + threadIdx.x;
    if (tid >= NRES * 1408) return;
    int n = tid / 1408, col = tid - n * 1408;
    float v;
    if (col < 128) {
        int h = col >> 4, d = col & 15;
        v = res_s[((size_t)h * NRES + n) * 16 + d];
    } else if (col < 320) {
        int idx = col - 128; int h = idx / 24, r = idx % 24;
        v = rpl[((size_t)h * NRES + n) * 24 + r];
    } else if (col < 384) {
        int idx = col - 320; int h = idx >> 3, d = idx & 7;
        v = rpn[((size_t)h * NRES + n) * 8 + d];
    } else {
        int idx = col - 384; int h = idx >> 7, d = idx & 127;
        v = rpt[((size_t)n * 128 + d) * 16 + h];
    }
    of[tid] = f2bf(v);
}

// out = LN(x + resid) * g + b ; also bf16 copy
__global__ __launch_bounds__(128) void k_layernorm(const float* __restrict__ x,
        const float* __restrict__ resid, const float* __restrict__ g,
        const float* __restrict__ b, float* __restrict__ out, bf16_t* __restrict__ outbf) {
    __shared__ float s1[128], s2[128];
    int n = blockIdx.x, t = threadIdx.x;
    float v[3], sum = 0.f, sq = 0.f;
    #pragma unroll
    for (int r = 0; r < 3; ++r) {
        size_t idx = (size_t)n * DIMC + t + r * 128;
        float a = x[idx] + resid[idx];
        v[r] = a; sum += a; sq += a * a;
    }
    s1[t] = sum; s2[t] = sq; __syncthreads();
    for (int o = 64; o; o >>= 1) {
        if (t < o) { s1[t] += s1[t + o]; s2[t] += s2[t + o]; }
        __syncthreads();
    }
    float mu = s1[0] * (1.0f / DIMC);
    float var = s2[0] * (1.0f / DIMC) - mu * mu;
    float rstd = rsqrtf(var + 1e-5f);
    #pragma unroll
    for (int r = 0; r < 3; ++r) {
        int c = t + r * 128;
        float o = (v[r] - mu) * rstd * g[c] + b[c];
        out[(size_t)n * DIMC + c] = o;
        if (outbf) outbf[(size_t)n * DIMC + c] = f2bf(o);
    }
}

// 6D->rotation, coordinate update, write coors/rotations outputs.
__global__ void k_final(const float* __restrict__ steps, const float* __restrict__ rot_in,
                        const float* __restrict__ coors_in,
                        float* __restrict__ out_coors, float* __restrict__ out_rot) {
    int n = blockIdx.x * 256 + threadIdx.x;
    if (n >= NRES) return;
    const float* s = steps + (size_t)n * 16;
    float a1x = s[0], a1y = s[1], a1z = s[2];
    float a2x = s[3], a2y = s[4], a2z = s[5];
    float n1 = fmaxf(sqrtf(a1x*a1x + a1y*a1y + a1z*a1z), 1e-12f);
    float b1x = a1x/n1, b1y = a1y/n1, b1z = a1z/n1;
    float dp = b1x*a2x + b1y*a2y + b1z*a2z;
    float c2x = a2x - dp*b1x, c2y = a2y - dp*b1y, c2z = a2z - dp*b1z;
    float n2 = fmaxf(sqrtf(c2x*c2x + c2y*c2y + c2z*c2z), 1e-12f);
    float b2x = c2x/n2, b2y = c2y/n2, b2z = c2z/n2;
    float b3x = b1y*b2z - b1z*b2y;
    float b3y = b1z*b2x - b1x*b2z;
    float b3z = b1x*b2y - b1y*b2x;
    float U[9] = { b1x,b1y,b1z,  b2x,b2y,b2z,  b3x,b3y,b3z };
    float R[9];
    #pragma unroll
    for (int a = 0; a < 9; ++a) R[a] = rot_in[(size_t)n * 9 + a];
    // crd_update[r] = sum_c s[6+c]*R[c][r]
    #pragma unroll
    for (int r = 0; r < 3; ++r)
        out_coors[(size_t)n * 3 + r] = coors_in[(size_t)n * 3 + r]
            + s[6]*R[0*3+r] + s[7]*R[1*3+r] + s[8]*R[2*3+r];
    // new_rot = R @ U
    #pragma unroll
    for (int r = 0; r < 3; ++r)
        #pragma unroll
        for (int cc = 0; cc < 3; ++cc)
            out_rot[(size_t)n * 9 + r*3 + cc] =
                R[r*3+0]*U[0*3+cc] + R[r*3+1]*U[1*3+cc] + R[r*3+2]*U[2*3+cc];
}

// ---------------------------------------------------------------------------
extern "C" void kernel_launch(void* const* d_in, const int* in_sizes, int n_in,
                              void* d_out, int out_size, void* d_ws, size_t ws_size,
                              hipStream_t stream) {
    (void)in_sizes; (void)n_in; (void)out_size; (void)ws_size;
    const float* nodes = (const float*)d_in[0];
    const float* edges = (const float*)d_in[1];
    const float* coors = (const float*)d_in[2];
    const float* rots  = (const float*)d_in[3];
    const unsigned char* mask = (const unsigned char*)d_in[4];
    const float* Wq_s = (const float*)d_in[5];
    const float* Wk_s = (const float*)d_in[6];
    const float* Wv_s = (const float*)d_in[7];
    const float* Wq_p = (const float*)d_in[8];
    const float* Wk_p = (const float*)d_in[9];
    const float* Wv_p = (const float*)d_in[10];
    const float* pw   = (const float*)d_in[11];
    const float* Wpair= (const float*)d_in[12];
    const float* bpair= (const float*)d_in[13];
    const float* Wout = (const float*)d_in[14];
    const float* bout = (const float*)d_in[15];
    const float* g1   = (const float*)d_in[16];
    const float* b1   = (const float*)d_in[17];
    const float* Wf1  = (const float*)d_in[18];
    const float* bf1  = (const float*)d_in[19];
    const float* Wf2  = (const float*)d_in[20];
    const float* bf2  = (const float*)d_in[21];
    const float* Wf3  = (const float*)d_in[22];
    const float* bf3  = (const float*)d_in[23];
    const float* g2   = (const float*)d_in[24];
    const float* b2   = (const float*)d_in[25];
    const float* Wproj= (const float*)d_in[26];
    const float* bproj= (const float*)d_in[27];

    float* out_x = (float*)d_out;
    float* out_coors = out_x + NRES * DIMC;
    float* out_rot   = out_coors + NRES * 3;

    size_t off = 0;
    auto alloc = [&](size_t bytes) -> void* {
        off = (off + 255) & ~(size_t)255;
        void* p = (char*)d_ws + off;
        off += bytes;
        return p;
    };
    bf16_t* xbf   = (bf16_t*)alloc((size_t)NRES * DIMC * 2);
    bf16_t* WqsT  = (bf16_t*)alloc((size_t)128 * DIMC * 2);
    bf16_t* WksT  = (bf16_t*)alloc((size_t)128 * DIMC * 2);
    bf16_t* WvsT  = (bf16_t*)alloc((size_t)128 * DIMC * 2);
    bf16_t* WqpT  = (bf16_t*)alloc((size_t)96  * DIMC * 2);
    bf16_t* WkpT  = (bf16_t*)alloc((size_t)96  * DIMC * 2);
    bf16_t* WvpT  = (bf16_t*)alloc((size_t)192 * DIMC * 2);
    bf16_t* WoutT = (bf16_t*)alloc((size_t)DIMC * 1408 * 2);
    bf16_t* Wf1T  = (bf16_t*)alloc((size_t)DIMC * DIMC * 2);
    bf16_t* Wf2T  = (bf16_t*)alloc((size_t)DIMC * DIMC * 2);
    bf16_t* Wf3T  = (bf16_t*)alloc((size_t)DIMC * DIMC * 2);
    bf16_t* WprT  = (bf16_t*)alloc((size_t)16 * DIMC * 2);
    float*  bpr16 = (float*)alloc(16 * 4);
    float*  qs    = (float*)alloc((size_t)NRES * 128 * 4);
    float*  ks    = (float*)alloc((size_t)NRES * 128 * 4);
    float*  vs    = (float*)alloc((size_t)NRES * 128 * 4);
    float*  qpr   = (float*)alloc((size_t)NRES * 96 * 4);
    float*  kpr   = (float*)alloc((size_t)NRES * 96 * 4);
    float*  vpr   = (float*)alloc((size_t)NRES * 192 * 4);
    bf16_t* qsb   = (bf16_t*)alloc((size_t)HNUM * NRES * 32 * 2);
    bf16_t* ksb   = (bf16_t*)alloc((size_t)HNUM * NRES * 32 * 2);
    bf16_t* vsT   = (bf16_t*)alloc((size_t)HNUM * 16 * NRES * 2);
    float*  qp    = (float*)alloc((size_t)HNUM * NRES * 12 * 4);
    float*  kp    = (float*)alloc((size_t)HNUM * NRES * 12 * 4);
    bf16_t* vpt   = (bf16_t*)alloc((size_t)HNUM * 32 * NRES * 2);
    float*  logits= (float*)alloc((size_t)HNUM * NRES * NRES * 4);
    bf16_t* attnb = (bf16_t*)alloc((size_t)16 * NRES * NRES * 2); // 16 h-rows: 8..15 garbage pads
    float*  res_s = (float*)alloc((size_t)HNUM * NRES * 16 * 4);
    float*  rpg   = (float*)alloc((size_t)HNUM * NRES * 32 * 4);
    float*  rpt   = (float*)alloc((size_t)NRES * 128 * 16 * 4);
    float*  rpl   = (float*)alloc((size_t)HNUM * NRES * 24 * 4);
    float*  rpn   = (float*)alloc((size_t)HNUM * NRES * 8 * 4);
    bf16_t* ofbf  = (bf16_t*)alloc((size_t)NRES * 1408 * 2);
    float*  aout  = (float*)alloc((size_t)NRES * DIMC * 4);
    float*  x1    = (float*)alloc((size_t)NRES * DIMC * 4);
    bf16_t* x1bf  = (bf16_t*)alloc((size_t)NRES * DIMC * 2);
    float*  h1    = (float*)alloc((size_t)NRES * DIMC * 4);
    bf16_t* h1bf  = (bf16_t*)alloc((size_t)NRES * DIMC * 2);
    float*  h2    = (float*)alloc((size_t)NRES * DIMC * 4);
    bf16_t* h2bf  = (bf16_t*)alloc((size_t)NRES * DIMC * 2);
    float*  f3    = (float*)alloc((size_t)NRES * DIMC * 4);
    bf16_t* x2bf  = (bf16_t*)alloc((size_t)NRES * DIMC * 2);
    float*  steps = (float*)alloc((size_t)NRES * 16 * 4);

    auto blocks = [](long long n) { return dim3((unsigned)((n + 255) / 256)); };
    auto gemm = [&](const bf16_t* A, long long ars, long long abs_,
                    const bf16_t* Bt, long long brs, long long bbs,
                    float* C, long long crs, long long cbs, bf16_t* Cbf,
                    const float* bias, int M, int N, int K,
                    float alpha, int acc, int relu, int batches) {
        int tiles = (M / 16) * (N / 16);
        dim3 grid((tiles + 3) / 4, batches);
        k_gemm_bf16<<<grid, 128, 0, stream>>>(A, ars, abs_, Bt, brs, bbs,
                                              C, crs, cbs, Cbf, bias,
                                              M, N, K, alpha, acc, relu);
    };

    // --- conversions ---
    k_f32_to_bf16<<<blocks((long long)NRES * DIMC), 256, 0, stream>>>(nodes, xbf, NRES * DIMC);
    k_wtrans<<<blocks((long long)128 * DIMC), 256, 0, stream>>>(Wq_s, WqsT, DIMC, 128, 128);
    k_wtrans<<<blocks((long long)128 * DIMC), 256, 0, stream>>>(Wk_s, WksT, DIMC, 128, 128);
    k_wtrans<<<blocks((long long)128 * DIMC), 256, 0, stream>>>(Wv_s, WvsT, DIMC, 128, 128);
    k_wtrans<<<blocks((long long)96 * DIMC), 256, 0, stream>>>(Wq_p, WqpT, DIMC, 96, 96);
    k_wtrans<<<blocks((long long)96 * DIMC), 256, 0, stream>>>(Wk_p, WkpT, DIMC, 96, 96);
    k_wtrans<<<blocks((long long)192 * DIMC), 256, 0, stream>>>(Wv_p, WvpT, DIMC, 192, 192);
    k_wtrans<<<blocks((long long)DIMC * 1408), 256, 0, stream>>>(Wout, WoutT, 1408, DIMC, DIMC);
    k_wtrans<<<blocks((long long)DIMC * DIMC), 256, 0, stream>>>(Wf1, Wf1T, DIMC, DIMC, DIMC);
    k_wtrans<<<blocks((long long)DIMC * DIMC), 256, 0, stream>>>(Wf2, Wf2T, DIMC, DIMC, DIMC);
    k_wtrans<<<blocks((long long)DIMC * DIMC), 256, 0, stream>>>(Wf3, Wf3T, DIMC, DIMC, DIMC);
    k_wtrans<<<blocks((long long)16 * DIMC), 256, 0, stream>>>(Wproj, WprT, DIMC, 9, 16);
    k_padbias<<<1, 16, 0, stream>>>(bproj, bpr16, 9, 16);

    // --- QKV projections (WMMA) ---
    gemm(xbf, DIMC, 0, WqsT, DIMC, 0, qs, 128, 0, nullptr, nullptr, NRES, 128, DIMC, 1.f, 0, 0, 1);
    gemm(xbf, DIMC, 0, WksT, DIMC, 0, ks, 128, 0, nullptr, nullptr, NRES, 128, DIMC, 1.f, 0, 0, 1);
    gemm(xbf, DIMC, 0, WvsT, DIMC, 0, vs, 128, 0, nullptr, nullptr, NRES, 128, DIMC, 1.f, 0, 0, 1);
    gemm(xbf, DIMC, 0, WqpT, DIMC, 0, qpr, 96, 0, nullptr, nullptr, NRES, 96, DIMC, 1.f, 0, 0, 1);
    gemm(xbf, DIMC, 0, WkpT, DIMC, 0, kpr, 96, 0, nullptr, nullptr, NRES, 96, DIMC, 1.f, 0, 0, 1);
    gemm(xbf, DIMC, 0, WvpT, DIMC, 0, vpr, 192, 0, nullptr, nullptr, NRES, 192, DIMC, 1.f, 0, 0, 1);

    k_pack_qkv<<<blocks((long long)HNUM * NRES * 32), 256, 0, stream>>>(qs, ks, vs, qsb, ksb, vsT);
    k_points<<<blocks(NRES), 256, 0, stream>>>(qpr, kpr, vpr, rots, coors, qp, kp, vpt);

    // --- logits: pair init (edges pass 1), + QK WMMA, - point distances ---
    k_pair_logits<<<dim3(NRES / 64, NRES), 256, 0, stream>>>(edges, Wpair, bpair, logits);
    gemm(qsb, 32, (long long)NRES * 32, ksb, 32, (long long)NRES * 32,
         logits, NRES, (long long)NRES * NRES, nullptr, nullptr,
         NRES, NRES, 32, SCALAR_SCALE_C, 1, 0, HNUM);
    k_point_logits<<<dim3(3, NRES, HNUM), 256, 0, stream>>>(qp, kp, pw, logits);
    k_softmax<<<dim3(NRES, HNUM), 256, 0, stream>>>(logits, mask, attnb);

    // --- attention outputs ---
    gemm(attnb, NRES, (long long)NRES * NRES, vsT, NRES, (long long)16 * NRES,
         res_s, 16, (long long)NRES * 16, nullptr, nullptr, NRES, 16, NRES, 1.f, 0, 0, HNUM);
    gemm(attnb, NRES, (long long)NRES * NRES, vpt, NRES, (long long)32 * NRES,
         rpg, 32, (long long)NRES * 32, nullptr, nullptr, NRES, 32, NRES, 1.f, 0, 0, HNUM);
    k_res_pair<<<NRES, 256, 0, stream>>>(edges, attnb, rpt);   // edges pass 2 (WMMA)
    k_resp_local<<<blocks((long long)HNUM * NRES), 256, 0, stream>>>(rpg, rots, coors, rpl, rpn);
    k_concat<<<blocks((long long)NRES * 1408), 256, 0, stream>>>(res_s, rpl, rpn, rpt, ofbf);

    // --- output projection, residual + LN, FFN, residual + LN ---
    gemm(ofbf, 1408, 0, WoutT, 1408, 0, aout, DIMC, 0, nullptr, bout,
         NRES, DIMC, 1408, 1.f, 0, 0, 1);
    k_layernorm<<<NRES, 128, 0, stream>>>(aout, nodes, g1, b1, x1, x1bf);
    gemm(x1bf, DIMC, 0, Wf1T, DIMC, 0, h1, DIMC, 0, h1bf, bf1, NRES, DIMC, DIMC, 1.f, 0, 1, 1);
    gemm(h1bf, DIMC, 0, Wf2T, DIMC, 0, h2, DIMC, 0, h2bf, bf2, NRES, DIMC, DIMC, 1.f, 0, 1, 1);
    gemm(h2bf, DIMC, 0, Wf3T, DIMC, 0, f3, DIMC, 0, nullptr, bf3, NRES, DIMC, DIMC, 1.f, 0, 0, 1);
    k_layernorm<<<NRES, 128, 0, stream>>>(f3, x1, g2, b2, out_x, x2bf);

    // --- rigid update ---
    gemm(x2bf, DIMC, 0, WprT, DIMC, 0, steps, 16, 0, nullptr, bpr16,
         NRES, 16, DIMC, 1.f, 0, 0, 1);
    k_final<<<blocks(NRES), 256, 0, stream>>>(steps, rots, coors, out_coors, out_rot);
}
